// RelPositionAttend_3762391352015
// MI455X (gfx1250) — matrix operand order repaired and checked by reading, not compiled, not run on hardware
//
#include <hip/hip_runtime.h>
#include <hip/hip_bf16.h>
#include <stdint.h>

typedef __attribute__((ext_vector_type(16))) _Float16 v16h;
typedef __attribute__((ext_vector_type(8)))  _Float16 v8h;
typedef __attribute__((ext_vector_type(8)))  float    v8f;

#define BB 4
#define HH 16
#define TT 1024
#define DD 64
#define MM 1024
#define PP 2047
#define PPAD (PP + 32)   // 16 zero rows of padding front and back

// ---------------- WMMA fragment helpers (wave32, 16x16x32 f16) ----------------
// A (16x32 f16): lane L holds row m=L%16; h2=L/16 selects K halves:
//   vector elems e = A[m][h2*16 + e] (16 contiguous K values)
// B (32x16 f16): lane L holds col n=L%16; elems e = B[(L/16)*16 + e][n]
//   i.e. 16 contiguous K values when B^T (col-major B) is stored row-major.
// C/D (16x16 f32): lane L holds col n=L%16, rows 8*(L/16)+v for v=0..7.

static __device__ __forceinline__ v16h load_afrag_f16(const _Float16* rowbase, int h2) {
  v8h lo = *(const v8h*)(rowbase + h2 * 16);
  v8h hi = *(const v8h*)(rowbase + h2 * 16 + 8);
  v16h r;
#pragma unroll
  for (int i = 0; i < 8; ++i) { r[i] = lo[i]; r[i + 8] = hi[i]; }
  return r;
}

static __device__ __forceinline__ v16h load_bfrag_f16(const _Float16* p) {
  return *(const v16h*)p;
}

static __device__ __forceinline__ v8f wmma_f16(v16h a, v16h b, v8f c) {
  return __builtin_amdgcn_wmma_f32_16x16x32_f16(false, a, false, b, (short)0, c,
                                                false, false);
}

// LDS byte offset for inline asm (generic LDS addr = {aperture, offset[31:0]})
static __device__ __forceinline__ uint32_t lds_off(const void* p) {
  return (uint32_t)(uintptr_t)p;
}

// ---------------- Kernel 0: fp32 -> f16 bulk convert ----------------
__global__ __launch_bounds__(256) void cvt_kernel(const float* __restrict__ src,
                                                  _Float16* __restrict__ dst, int n) {
  int i = blockIdx.x * blockDim.x + threadIdx.x;
  if (i < n) dst[i] = (_Float16)src[i];
}

// ---------------- Kernel 1: prep q+u, q+v, k (f16 row-major) and v^T ----------------
__global__ __launch_bounds__(256) void prep_kernel(
    const float* __restrict__ q, const float* __restrict__ k,
    const float* __restrict__ v, const float* __restrict__ ub,
    const float* __restrict__ vb, _Float16* __restrict__ qu16,
    _Float16* __restrict__ qv16, _Float16* __restrict__ k16,
    _Float16* __restrict__ vT16) {
  int idx = blockIdx.x * blockDim.x + threadIdx.x;  // over B*H*T*D
  int d  = idx & (DD - 1);
  int t  = (idx >> 6) & (TT - 1);
  int bh = idx >> 16;               // /(T*D)
  int h  = bh & (HH - 1);
  float qe = q[idx];
  qu16[idx] = (_Float16)(qe + ub[h * DD + d]);
  qv16[idx] = (_Float16)(qe + vb[h * DD + d]);
  k16[idx]  = (_Float16)k[idx];
  vT16[(size_t)bh * TT * DD + (size_t)d * TT + t] = (_Float16)v[idx];
}

// ---------------- Kernel 2: p = pos_emb @ W_pos^T -> f16 [H][PPAD][D] ----------------
__global__ __launch_bounds__(128) void posproj_kernel(
    const _Float16* __restrict__ pe16, const _Float16* __restrict__ wp16,
    _Float16* __restrict__ p16) {
  int lane = threadIdx.x & 31;
  int wave = threadIdx.x >> 5;
  int lm = lane & 15, h2 = lane >> 4;
  int j0 = blockIdx.x * 64 + wave * 16;  // rows of pos (P dim)
  int n0 = blockIdx.y * 64;              // cols (M dim)

  int jrow = j0 + lm; if (jrow >= PP) jrow = PP - 1;
  const _Float16* arow = pe16 + (size_t)jrow * MM;

  v8f acc[4] = {};
  for (int m0 = 0; m0 < MM; m0 += 32) {
    v16h afrag = load_afrag_f16(arow + m0, h2);
#pragma unroll
    for (int t = 0; t < 4; ++t) {
      int n = n0 + t * 16 + lm;
      v16h bfrag = load_bfrag_f16(wp16 + (size_t)n * MM + m0 + h2 * 16);
      acc[t] = wmma_f16(afrag, bfrag, acc[t]);
    }
  }
#pragma unroll
  for (int t = 0; t < 4; ++t) {
    int n = n0 + t * 16 + lm;
    int h = n >> 6, d = n & 63;
#pragma unroll
    for (int vv = 0; vv < 8; ++vv) {
      int j = j0 + (h2 << 3) + vv;
      if (j < PP)
        p16[((size_t)h * PPAD + (j + 16)) * DD + d] = (_Float16)acc[t][vv];
    }
  }
}

// ---------------- Kernel 3: fused flash attention with rel-shift ----------------
// bd[q,k] = bd_raw[q, (T-1)+k-q]; per (16q x 32k) tile we compute a 48-wide
// strip of bd_raw with 3 WMMA n-tiles and gather diagonals through LDS.
// K/V tiles (shared by all 4 waves) are staged into LDS with gfx1250 async
// global->LDS copies, overlapped with the bd-strip WMMAs.
__global__ __launch_bounds__(128) void attn_kernel(
    const _Float16* __restrict__ qu, const _Float16* __restrict__ qv,
    const _Float16* __restrict__ k16, const _Float16* __restrict__ vT,
    const _Float16* __restrict__ p16, const uint8_t* __restrict__ mask,
    float* __restrict__ out) {
  __shared__ float    s_strip[4][16][48];
  __shared__ float    s_stile[4][16][32];
  __shared__ _Float16 s_probs[4][16][32];
  __shared__ float    s_row[4][16];
  __shared__ __align__(32) _Float16 s_ktile[32][64];  // K rows x D (4 KB)
  __shared__ __align__(32) _Float16 s_vtile[64][32];  // D rows x K (4 KB)

  int tid  = threadIdx.x;
  int lane = tid & 31;
  int wave = tid >> 5;
  int lm = lane & 15, h2 = lane >> 4;
  int bh = blockIdx.y;
  int b  = bh >> 4;
  int h  = bh & (HH - 1);
  int q0 = blockIdx.x * 64 + wave * 16;

  const _Float16* qu_row = qu + ((size_t)bh * TT + q0 + lm) * DD;
  const _Float16* qv_row = qv + ((size_t)bh * TT + q0 + lm) * DD;
  const _Float16* kbh = k16 + (size_t)bh * TT * DD;
  const _Float16* vbh = vT + (size_t)bh * TT * DD;
  const _Float16* ph  = p16 + (size_t)h * PPAD * DD;
  const uint8_t*  mrow = mask + (size_t)b * TT;

  // A fragments for this wave's 16 q rows (K = 64 -> 2 K-steps), loaded once
  v16h aqu0 = load_afrag_f16(qu_row, h2);
  v16h aqu1 = load_afrag_f16(qu_row + 32, h2);
  v16h aqv0 = load_afrag_f16(qv_row, h2);
  v16h aqv1 = load_afrag_f16(qv_row + 32, h2);

  v8f O[4] = {};
  float run_m = -3.0e38f, run_l = 0.f;

  for (int k0 = 0; k0 < TT; k0 += 32) {
    // ---- stage K tile (32x64 f16) and V tile (64x32 f16) into LDS, async.
    // 256 16-byte chunks each; 128 threads x 2 rounds. ASYNCcnt-tracked.
#pragma unroll
    for (int r = 0; r < 2; ++r) {
      int c = tid + r * 128;
      int krow = c >> 3, koff = (c & 7) * 16;  // 8 chunks per 128B K row
      const char* gk = (const char*)(kbh + (size_t)(k0 + krow) * DD) + koff;
      uint32_t lk = lds_off((const char*)&s_ktile[krow][0] + koff);
      asm volatile("global_load_async_to_lds_b128 %0, %1, off"
                   :: "v"(lk), "v"(gk) : "memory");
      int vrow = c >> 2, voff = (c & 3) * 16;  // 4 chunks per 64B V row
      const char* gv = (const char*)(vbh + (size_t)vrow * TT + k0) + voff;
      uint32_t lv = lds_off((const char*)&s_vtile[vrow][0] + voff);
      asm volatile("global_load_async_to_lds_b128 %0, %1, off"
                   :: "v"(lv), "v"(gv) : "memory");
    }

    // ---- bd strip (overlaps the async copies):
    // padded j index jp = (T-1)+k0-q0 + c - 16, c in [0,48)
    int jpbase = (TT - 1) + k0 - q0;
    v8f bd[3] = {{}, {}, {}};
#pragma unroll
    for (int t = 0; t < 3; ++t) {
      int jp = jpbase + (t - 1) * 16 + lm;
      const _Float16* prow = ph + (size_t)jp * DD + h2 * 16;
      bd[t] = wmma_f16(aqv0, load_bfrag_f16(prow), bd[t]);
      bd[t] = wmma_f16(aqv1, load_bfrag_f16(prow + 32), bd[t]);
    }
#pragma unroll
    for (int t = 0; t < 3; ++t)
#pragma unroll
      for (int vv = 0; vv < 8; ++vv)
        s_strip[wave][(h2 << 3) + vv][t * 16 + lm] = bd[t][vv];

    asm volatile("s_wait_asynccnt 0" ::: "memory");  // own async writes landed
    __syncthreads();                                 // cross-wave visibility

    // ---- ac tiles from LDS K tile, combine with strip diagonal, scale + mask
#pragma unroll
    for (int t = 0; t < 2; ++t) {
      const _Float16* krow = &s_ktile[t * 16 + lm][h2 * 16];
      v8f ac = {};
      ac = wmma_f16(aqu0, load_bfrag_f16(krow), ac);
      ac = wmma_f16(aqu1, load_bfrag_f16(krow + 32), ac);
#pragma unroll
      for (int vv = 0; vv < 8; ++vv) {
        int qr = (h2 << 3) + vv;
        int kc = t * 16 + lm;
        float bdv = s_strip[wave][qr][16 + kc - qr];  // rel_shift gather
        float sv = (ac[vv] + bdv) * 0.125f;           // 1/sqrt(D)
        if (!mrow[k0 + kc]) sv = -3.0e38f;
        s_stile[wave][qr][kc] = sv;
      }
    }
    __syncthreads();

    // ---- online softmax: 2 lanes per row, 16 cols each
    {
      int r = lane >> 1, half = lane & 1;
      float lmax = -3.0e38f;
#pragma unroll
      for (int c = 0; c < 16; ++c)
        lmax = fmaxf(lmax, s_stile[wave][r][half * 16 + c]);
      float tmax = fmaxf(lmax, __shfl_xor(lmax, 1));
      float nm = fmaxf(run_m, tmax);
      float corr = __expf(run_m - nm);
      float ls = 0.f;
#pragma unroll
      for (int c = 0; c < 16; ++c) {
        float e = __expf(s_stile[wave][r][half * 16 + c] - nm);
        s_probs[wave][r][half * 16 + c] = (_Float16)e;
        ls += e;
      }
      ls += __shfl_xor(ls, 1);
      run_l = run_l * corr + ls;
      run_m = nm;
      if (half == 0) s_row[wave][r] = corr;
    }
    __syncthreads();

    // ---- rescale O, accumulate P @ V from LDS V tile
    float cr[8];
#pragma unroll
    for (int vv = 0; vv < 8; ++vv) cr[vv] = s_row[wave][(h2 << 3) + vv];
#pragma unroll
    for (int t = 0; t < 4; ++t)
#pragma unroll
      for (int vv = 0; vv < 8; ++vv) O[t][vv] *= cr[vv];

    v16h pa = load_afrag_f16(&s_probs[wave][lm][0], h2);
#pragma unroll
    for (int t = 0; t < 4; ++t) {
      const _Float16* vrow = &s_vtile[t * 16 + lm][h2 * 16];
      O[t] = wmma_f16(pa, load_bfrag_f16(vrow), O[t]);
    }
    __syncthreads();
  }

  // normalize by row sums and store fp32 output (B,H,T,D)
  if ((lane & 1) == 0) s_row[wave][lane >> 1] = 1.0f / run_l;
  __syncthreads();
#pragma unroll
  for (int t = 0; t < 4; ++t)
#pragma unroll
    for (int vv = 0; vv < 8; ++vv) {
      int qr = (h2 << 3) + vv;
      out[((size_t)bh * TT + q0 + qr) * DD + t * 16 + lm] =
          O[t][vv] * s_row[wave][qr];
    }
}

extern "C" void kernel_launch(void* const* d_in, const int* in_sizes, int n_in,
                              void* d_out, int out_size, void* d_ws, size_t ws_size,
                              hipStream_t stream) {
  (void)in_sizes; (void)n_in; (void)out_size; (void)ws_size;
  const float*   q       = (const float*)d_in[0];
  const float*   k       = (const float*)d_in[1];
  const float*   v       = (const float*)d_in[2];
  const float*   pos_emb = (const float*)d_in[3];
  const uint8_t* mask    = (const uint8_t*)d_in[4];
  const float*   W_pos   = (const float*)d_in[5];
  const float*   ub      = (const float*)d_in[6];
  const float*   vb      = (const float*)d_in[7];

  const size_t S = (size_t)BB * HH * TT * DD;  // elements per f16 tensor
  const size_t p16_bytes = (size_t)HH * PPAD * DD * sizeof(_Float16);
  const size_t pe_elems  = (size_t)PP * MM;
  const size_t wp_elems  = (size_t)MM * MM;

  char* w = (char*)d_ws;
  _Float16* qu16 = (_Float16*)(w);
  _Float16* qv16 = (_Float16*)(w + 2 * S);
  _Float16* k16  = (_Float16*)(w + 4 * S);
  _Float16* vT16 = (_Float16*)(w + 6 * S);
  _Float16* p16  = (_Float16*)(w + 8 * S);
  _Float16* pe16 = (_Float16*)(w + 8 * S + p16_bytes);
  _Float16* wp16 = (_Float16*)(w + 8 * S + p16_bytes + 2 * pe_elems);

  hipMemsetAsync(p16, 0, p16_bytes, stream);  // zero-padded rel-pos table
  cvt_kernel<<<dim3((unsigned)((pe_elems + 255) / 256)), 256, 0, stream>>>(
      pos_emb, pe16, (int)pe_elems);
  cvt_kernel<<<dim3((unsigned)((wp_elems + 255) / 256)), 256, 0, stream>>>(
      W_pos, wp16, (int)wp_elems);
  prep_kernel<<<dim3((unsigned)(S / 256)), 256, 0, stream>>>(
      q, k, v, ub, vb, qu16, qv16, k16, vT16);
  posproj_kernel<<<dim3(32, 16), 128, 0, stream>>>(pe16, wp16, p16);
  attn_kernel<<<dim3(TT / 64, BB * HH), 128, 0, stream>>>(
      qu16, qv16, k16, vT16, p16, mask, (float*)d_out);
}
